// TSTEncoder_85787676770363
// MI455X (gfx1250) — compile-verified, matrix operand order checked by reading, not run on hardware
//
#include <hip/hip_runtime.h>
#include <hip/hip_bf16.h>
#include <math.h>

// ---------------------------------------------------------------------------
// TST encoder for MI455X (gfx1250): all GEMMs via v_wmma_f32_16x16x32_f16,
// f32 accumulation, double-buffered LDS tiles, all B operands pre-transposed
// so staging is pure 16B vector loads. wave32 / 8-wave workgroups.
// ---------------------------------------------------------------------------

typedef __attribute__((ext_vector_type(16))) _Float16 v16h;
typedef __attribute__((ext_vector_type(8)))  _Float16 v8h;
typedef __attribute__((ext_vector_type(8)))  float    v8f;

#define TL 4
#define TB 8
#define TS 512
#define TD 512
#define TH_ 8
#define TF 2048
#define TDK 64

#define BM 128
#define BN 64
#define BK 32

// ---------------------------------------------------------------------------
// f32 (K x N) -> f16 transposed (N x K), LDS-tiled, one 32x32 tile per block.
// Used once per weight family so every GEMM B operand is K-contiguous.
// ---------------------------------------------------------------------------
__global__ __launch_bounds__(256) void cvt_transpose_kernel(
    const float* __restrict__ in, _Float16* __restrict__ out, int K, int N) {
  __shared__ _Float16 tile[32][33];
  const long long z = blockIdx.z;
  in += z * (long long)K * N;
  out += z * (long long)K * N;
  const int nb = blockIdx.x * 32;
  const int kb = blockIdx.y * 32;
  const int tx = threadIdx.x & 31;
  const int ty = threadIdx.x >> 5;  // 0..7
#pragma unroll
  for (int i = 0; i < 32; i += 8)
    tile[ty + i][tx] = (_Float16)in[(long long)(kb + ty + i) * N + nb + tx];
  __syncthreads();
#pragma unroll
  for (int i = 0; i < 32; i += 8)
    out[(long long)(nb + ty + i) * K + kb + tx] = tile[tx][ty + i];
}

// ---------------------------------------------------------------------------
// LayerNorm over D=512, optional residual input, dual f32/f16 outputs.
// One block (256 threads = 8 waves) per row.
// ---------------------------------------------------------------------------
__device__ __forceinline__ float block_sum_512(float v, float* red) {
#pragma unroll
  for (int o = 16; o > 0; o >>= 1) v += __shfl_down(v, o, 32);
  int lane = threadIdx.x & 31;
  int w = threadIdx.x >> 5;
  if (lane == 0) red[w] = v;
  __syncthreads();
  float tot;
  if (threadIdx.x == 0) {
    tot = 0.f;
#pragma unroll
    for (int i = 0; i < 8; ++i) tot += red[i];
    red[0] = tot;
  }
  __syncthreads();
  tot = red[0];
  __syncthreads();
  return tot;
}

__global__ __launch_bounds__(256) void layernorm_kernel(
    const float* __restrict__ in, const float* __restrict__ res,
    const float* __restrict__ g, const float* __restrict__ b,
    float* __restrict__ outF, _Float16* __restrict__ outH) {
  __shared__ float red[8];
  __shared__ float rowv[TD];
  const long long r = blockIdx.x;
  const int t = threadIdx.x;
  const float* pin = in + r * TD;
  const float* pres = res ? res + r * TD : nullptr;

  float l = 0.f;
#pragma unroll
  for (int i = 0; i < 2; ++i) {
    int c = t + i * 256;
    float v = pin[c];
    if (pres) v += pres[c];
    rowv[c] = v;
    l += v;
  }
  float mean = block_sum_512(l, red) * (1.f / (float)TD);
  float lv = 0.f;
#pragma unroll
  for (int i = 0; i < 2; ++i) {
    int c = t + i * 256;
    float d = rowv[c] - mean;
    lv += d * d;
  }
  float var = block_sum_512(lv, red) * (1.f / (float)TD);
  float rstd = rsqrtf(var + 1e-5f);
#pragma unroll
  for (int i = 0; i < 2; ++i) {
    int c = t + i * 256;
    float y = (rowv[c] - mean) * rstd * g[c] + b[c];
    if (outF) outF[r * TD + c] = y;
    if (outH) outH[r * TD + c] = (_Float16)y;
  }
}

// ---------------------------------------------------------------------------
// RoPE, in place on (B,H,S,DK) f16. One row per thread (owns entire row).
// ---------------------------------------------------------------------------
__global__ __launch_bounds__(256) void rope_kernel(_Float16* __restrict__ x,
                                                   int nrows) {
  int row = blockIdx.x * blockDim.x + threadIdx.x;
  if (row >= nrows) return;
  int s = row % TS;
  _Float16* p = x + (long long)row * TDK;
  float x1[TDK / 2], x2[TDK / 2];
#pragma unroll
  for (int j = 0; j < TDK / 2; ++j) {
    x1[j] = (float)p[2 * j];
    x2[j] = (float)p[2 * j + 1];
  }
#pragma unroll
  for (int j = 0; j < TDK / 2; ++j) {
    float fr = __powf(10000.f, -(2.f * (float)j) / (float)TDK);
    float ang = (float)s * fr;
    float sn, cs;
    __sincosf(ang, &sn, &cs);
    p[j] = (_Float16)(x1[j] * cs - x2[j] * sn);
    p[j + TDK / 2] = (_Float16)(x2[j] * cs + x1[j] * sn);
  }
}

// ---------------------------------------------------------------------------
// Batched WMMA GEMM: D = alpha * (A @ Bt^T) [+bias] [+add] [gelu]
//   A : M x K row-major f16 (+ z*aBatch)
//   Bt: N x K row-major f16 (i.e. B pre-transposed; + z*bBatch)
//   scatter: 0 plain row-major (ldo)
//            1 (B*S,D) -> (B,H,S,DK)        (Q/K projections)
//            2 z=(b,h), q x dk -> (B,S,D)   (attn @ V output)
//            3 (B*S,D) -> (B,H,DK,S)        (V projection, K-contiguous for 2)
// Block: 256 threads (8 waves); tile 128x64x32; wave 32x32 (2x2 WMMA).
// Double-buffered LDS: global->regs prefetch overlaps the WMMA block,
// one barrier per K-step.
// ---------------------------------------------------------------------------
__global__ __launch_bounds__(256) void wmma_gemm_kernel(
    const _Float16* __restrict__ A, long long aBatch, int lda,
    const _Float16* __restrict__ Bt, long long bBatch, int ldb,
    const float* __restrict__ bias, const float* __restrict__ add,
    long long addBatch, float alpha, float* __restrict__ outF,
    _Float16* __restrict__ outH, long long oBatch, int ldo, int M, int N,
    int K, int gelu, int scatter) {
  __shared__ _Float16 lA[2][BM][BK];  // [buf][m][k]
  __shared__ _Float16 lB[2][BN][BK];  // [buf][n][k]

  const int tid = threadIdx.x;
  const int lane = tid & 31;
  const int wave = tid >> 5;
  const int wm = wave & 3;   // 4 waves along M
  const int wn = wave >> 2;  // 2 waves along N
  const int z = blockIdx.z;
  const int m0 = blockIdx.y * BM;
  const int n0 = blockIdx.x * BN;

  A += (long long)z * aBatch;
  Bt += (long long)z * bBatch;

  // staging coordinates: A 128x32 -> 16 halves/thread, B 64x32 -> 8/thread
  const int ra = tid >> 1;
  const int ca = (tid & 1) * 16;
  const int rb = tid >> 2;
  const int cb = (tid & 3) * 8;
  const _Float16* aRow = A + (long long)(m0 + ra) * lda;
  const _Float16* bRow = Bt + (long long)(n0 + rb) * ldb;

  v8f acc[2][2] = {};
  const int halfSel = lane >> 4;  // 16-bit A/B frag: K sub-block select
  const int kSub = halfSel * 8;
  const int lm = lane & 15;

  // prologue: stage tile 0
  v8h a0 = *(const v8h*)(aRow + ca);
  v8h a1 = *(const v8h*)(aRow + ca + 8);
  v8h b0 = *(const v8h*)(bRow + cb);
  *(v8h*)&lA[0][ra][ca] = a0;
  *(v8h*)&lA[0][ra][ca + 8] = a1;
  *(v8h*)&lB[0][rb][cb] = b0;
  __syncthreads();

  int cur = 0;
  for (int k0 = 0; k0 < K; k0 += BK) {
    const bool haveNext = (k0 + BK) < K;
    if (haveNext) {  // prefetch next tile into registers (overlaps WMMA)
      a0 = *(const v8h*)(aRow + k0 + BK + ca);
      a1 = *(const v8h*)(aRow + k0 + BK + ca + 8);
      b0 = *(const v8h*)(bRow + k0 + BK + cb);
      __builtin_prefetch(aRow + k0 + 2 * BK + ca, 0, 1);
    }

    // fragments (16-bit layout: lanes 0-15 K{0-7,16-23}, 16-31 K{8-15,24-31})
    v16h af[2], bf[2];
#pragma unroll
    for (int ti = 0; ti < 2; ++ti) {
      const _Float16* pa = &lA[cur][wm * 32 + ti * 16 + lm][kSub];
      v8h lo = *(const v8h*)pa;
      v8h hi = *(const v8h*)(pa + 16);
#pragma unroll
      for (int e = 0; e < 8; ++e) {
        af[ti][e] = lo[e];
        af[ti][e + 8] = hi[e];
      }
    }
#pragma unroll
    for (int tj = 0; tj < 2; ++tj) {
      const _Float16* pb = &lB[cur][wn * 32 + tj * 16 + lm][kSub];
      v8h lo = *(const v8h*)pb;
      v8h hi = *(const v8h*)(pb + 16);
#pragma unroll
      for (int e = 0; e < 8; ++e) {
        bf[tj][e] = lo[e];
        bf[tj][e + 8] = hi[e];
      }
    }
#pragma unroll
    for (int ti = 0; ti < 2; ++ti)
#pragma unroll
      for (int tj = 0; tj < 2; ++tj)
        acc[ti][tj] = __builtin_amdgcn_wmma_f32_16x16x32_f16(
            false, af[ti], false, bf[tj], (short)0, acc[ti][tj], false, false);

    if (haveNext) {  // write-side of the pipeline into the other buffer
      *(v8h*)&lA[cur ^ 1][ra][ca] = a0;
      *(v8h*)&lA[cur ^ 1][ra][ca + 8] = a1;
      *(v8h*)&lB[cur ^ 1][rb][cb] = b0;
    }
    __syncthreads();
    cur ^= 1;
  }

  // epilogue: 16x16 f32 D layout (VGPR e -> M = e (+8 for lanes 16-31))
#pragma unroll
  for (int ti = 0; ti < 2; ++ti) {
#pragma unroll
    for (int tj = 0; tj < 2; ++tj) {
#pragma unroll
      for (int e = 0; e < 8; ++e) {
        int m = m0 + wm * 32 + ti * 16 + e + (halfSel ? 8 : 0);
        int n = n0 + wn * 32 + tj * 16 + lm;
        float v = alpha * acc[ti][tj][e];
        if (bias) v += bias[n];
        if (add) v += add[(long long)z * addBatch + (long long)m * ldo + n];
        if (gelu) v = 0.5f * v * (1.0f + erff(v * 0.70710678118654752f));
        if (outF) outF[(long long)z * oBatch + (long long)m * ldo + n] = v;
        if (outH) {
          long long idx;
          if (scatter == 1) {  // (B*S, D) -> (B,H,S,DK)
            int b = m / TS, s = m - b * TS;
            int h = n / TDK, dk = n - h * TDK;
            idx = (((long long)(b * TH_ + h) * TS) + s) * TDK + dk;
          } else if (scatter == 2) {  // z=(b,h), rows=q, cols=dk -> (B,S,D)
            int b = z / TH_, h = z - b * TH_;
            idx = ((long long)(b * TS + m) * TD) + h * TDK + n;
          } else if (scatter == 3) {  // (B*S, D) -> (B,H,DK,S)
            int b = m / TS, s = m - b * TS;
            int h = n / TDK, dk = n - h * TDK;
            idx = (((long long)(b * TH_ + h) * TDK) + dk) * TS + s;
          } else {
            idx = (long long)z * oBatch + (long long)m * ldo + n;
          }
          outH[idx] = (_Float16)v;
        }
      }
    }
  }
}

// ---------------------------------------------------------------------------
// Talking-heads mix + softmax, fused per (b,q):
//   mixed[g] = sum_h pre[h,g]*raw[h]   (written back to sc as next `prev`)
//   soft     = softmax_k(mixed)        (one wave per head row)
//   attn[g]  = sum_h post[h,g]*soft[h] (f16 out for attn@V WMMA GEMM)
// ---------------------------------------------------------------------------
__global__ __launch_bounds__(256) void mix_softmax_kernel(
    float* __restrict__ sc, const float* __restrict__ pre,
    const float* __restrict__ post, _Float16* __restrict__ attn) {
  __shared__ float sraw[TH_][TS];
  __shared__ float smix[TH_][TS];
  const int bq = blockIdx.x;
  const int b = bq / TS, q = bq - b * TS;
  const int t = threadIdx.x;
  const long long base = (long long)b * TH_ * TS * TS + (long long)q * TS;

  for (int h = 0; h < TH_; ++h)
    for (int i = t; i < TS; i += 256)
      sraw[h][i] = sc[base + (long long)h * TS * TS + i];
  __syncthreads();

  for (int g = 0; g < TH_; ++g)
    for (int i = t; i < TS; i += 256) {
      float a = 0.f;
#pragma unroll
      for (int h = 0; h < TH_; ++h) a += pre[h * TH_ + g] * sraw[h][i];
      smix[g][i] = a;
      sc[base + (long long)g * TS * TS + i] = a;  // prev for next layer
    }
  __syncthreads();

  {  // softmax: wave w owns head row w (8 waves == 8 heads)
    int w = t >> 5, lane = t & 31;
    float mx = -3.4e38f;
    for (int i = lane; i < TS; i += 32) mx = fmaxf(mx, smix[w][i]);
#pragma unroll
    for (int o = 16; o > 0; o >>= 1) mx = fmaxf(mx, __shfl_xor(mx, o, 32));
    float sum = 0.f;
    for (int i = lane; i < TS; i += 32) {
      float e = __expf(smix[w][i] - mx);
      smix[w][i] = e;
      sum += e;
    }
#pragma unroll
    for (int o = 16; o > 0; o >>= 1) sum += __shfl_xor(sum, o, 32);
    float inv = 1.f / sum;
    for (int i = lane; i < TS; i += 32) smix[w][i] *= inv;
  }
  __syncthreads();

  for (int g = 0; g < TH_; ++g)
    for (int i = t; i < TS; i += 256) {
      float a = 0.f;
#pragma unroll
      for (int h = 0; h < TH_; ++h) a += post[h * TH_ + g] * smix[h][i];
      attn[base + (long long)g * TS * TS + i] = (_Float16)a;
    }
}

// ---------------------------------------------------------------------------
// Host orchestration
// ---------------------------------------------------------------------------
static inline dim3 gemm_grid(int M, int N, int batches) {
  return dim3((N + BN - 1) / BN, (M + BM - 1) / BM, batches);
}

extern "C" void kernel_launch(void* const* d_in, const int* in_sizes, int n_in,
                              void* d_out, int out_size, void* d_ws,
                              size_t ws_size, hipStream_t stream) {
  (void)in_sizes; (void)n_in; (void)out_size; (void)ws_size;
  const float* src    = (const float*)d_in[0];
  const float* ln_q_g = (const float*)d_in[1];
  const float* ln_q_b = (const float*)d_in[2];
  const float* ln_k_g = (const float*)d_in[3];
  const float* ln_k_b = (const float*)d_in[4];
  const float* ln_v_g = (const float*)d_in[5];
  const float* ln_v_b = (const float*)d_in[6];
  const float* Wq     = (const float*)d_in[7];
  const float* bq     = (const float*)d_in[8];
  const float* Wk     = (const float*)d_in[9];
  const float* bk     = (const float*)d_in[10];
  const float* Wv     = (const float*)d_in[11];
  const float* bv     = (const float*)d_in[12];
  const float* th_pre = (const float*)d_in[13];
  const float* th_post= (const float*)d_in[14];
  const float* Wo     = (const float*)d_in[15];
  const float* bo     = (const float*)d_in[16];
  const float* ln1_g  = (const float*)d_in[17];
  const float* ln1_b  = (const float*)d_in[18];
  const float* ln2_g  = (const float*)d_in[19];
  const float* ln2_b  = (const float*)d_in[20];
  const float* W1     = (const float*)d_in[21];
  const float* b1     = (const float*)d_in[22];
  const float* W2     = (const float*)d_in[23];
  const float* b2     = (const float*)d_in[24];

  const long long nTok = (long long)TB * TS;       // 4096
  const long long nDD  = (long long)TL * TD * TD;  // per family
  const long long nDF  = (long long)TL * TD * TF;
  const long long nBSD = nTok * TD;
  const long long nSc  = (long long)TB * TH_ * TS * TS;

  char* ws = (char*)d_ws;
  size_t off = 0;
  auto alloc = [&](size_t bytes) -> char* {
    char* p = ws + off;
    off = (off + bytes + 255) & ~(size_t)255;
    return p;
  };
  _Float16* wq16  = (_Float16*)alloc(nDD * 2);  // all weights stored N x K
  _Float16* wk16  = (_Float16*)alloc(nDD * 2);
  _Float16* wv16  = (_Float16*)alloc(nDD * 2);
  _Float16* wo16  = (_Float16*)alloc(nDD * 2);
  _Float16* w1_16 = (_Float16*)alloc(nDF * 2);
  _Float16* w2_16 = (_Float16*)alloc(nDF * 2);
  float*    x32   = (float*)   alloc(nBSD * 4);
  _Float16* x16   = (_Float16*)alloc(nBSD * 2);
  _Float16* ln16  = (_Float16*)alloc(nBSD * 2);
  _Float16* q16   = (_Float16*)alloc(nBSD * 2);  // (B,H,S,DK)
  _Float16* k16   = (_Float16*)alloc(nBSD * 2);  // (B,H,S,DK)
  _Float16* v16   = (_Float16*)alloc(nBSD * 2);  // (B,H,DK,S)  transposed
  float*    sc0   = (float*)   alloc(nSc * 4);
  float*    sc1   = (float*)   alloc(nSc * 4);
  _Float16* at16  = (_Float16*)alloc(nSc * 2);
  _Float16* ob16  = (_Float16*)alloc(nBSD * 2);
  float*    op32  = (float*)   alloc(nBSD * 4);
  _Float16* ff16  = (_Float16*)alloc((long long)nTok * TF * 2);
  float*    ff32  = (float*)   alloc(nBSD * 4);

  // ---- convert + transpose weights once: (K x N) f32 -> (N x K) f16 ----
  auto cvtT = [&](const float* s, _Float16* d, int K, int N) {
    cvt_transpose_kernel<<<dim3(N / 32, K / 32, TL), 256, 0, stream>>>(s, d, K,
                                                                       N);
  };
  cvtT(Wq, wq16, TD, TD);
  cvtT(Wk, wk16, TD, TD);
  cvtT(Wv, wv16, TD, TD);
  cvtT(Wo, wo16, TD, TD);
  cvtT(W1, w1_16, TD, TF);  // -> (F x D)
  cvtT(W2, w2_16, TF, TD);  // -> (D x F)

  const float scale = 1.0f / 8.0f;  // DK^-0.5
  const float* xr = src;

  for (int i = 0; i < TL; ++i) {
    float* cur  = (i & 1) ? sc1 : sc0;
    float* prev = (i & 1) ? sc0 : sc1;
    const long long iDD = (long long)i * TD * TD;
    const long long iDF = (long long)i * TD * TF;

    // Q/K/V: LN -> WMMA GEMM (+bias); Q,K -> (B,H,S,DK); V -> (B,H,DK,S)
    struct Proj {
      const float *g, *b, *bias;
      _Float16 *wgt, *dst;
      int scat;
    } proj[3] = {
        {ln_q_g + i * TD, ln_q_b + i * TD, bq + i * TD, wq16, q16, 1},
        {ln_k_g + i * TD, ln_k_b + i * TD, bk + i * TD, wk16, k16, 1},
        {ln_v_g + i * TD, ln_v_b + i * TD, bv + i * TD, wv16, v16, 3}};
    for (int p = 0; p < 3; ++p) {
      layernorm_kernel<<<(unsigned)nTok, 256, 0, stream>>>(
          xr, nullptr, proj[p].g, proj[p].b, nullptr, ln16);
      wmma_gemm_kernel<<<gemm_grid((int)nTok, TD, 1), 256, 0, stream>>>(
          ln16, 0, TD, proj[p].wgt + iDD, 0, TD, proj[p].bias, nullptr, 0,
          1.0f, nullptr, proj[p].dst, 0, TD, (int)nTok, TD, TD, 0,
          proj[p].scat);
    }
    rope_kernel<<<(TB * TH_ * TS) / 256, 256, 0, stream>>>(q16, TB * TH_ * TS);
    rope_kernel<<<(TB * TH_ * TS) / 256, 256, 0, stream>>>(k16, TB * TH_ * TS);

    // scores = scale * q @ k^T (+ prev), batched over (b,h); f32 out
    wmma_gemm_kernel<<<gemm_grid(TS, TS, TB * TH_), 256, 0, stream>>>(
        q16, (long long)TS * TDK, TDK, k16, (long long)TS * TDK, TDK, nullptr,
        (i > 0) ? prev : nullptr, (long long)TS * TS, scale, cur, nullptr,
        (long long)TS * TS, TS, TS, TS, TDK, 0, 0);

    // talking-heads mix + softmax; writes mixed back into `cur` (next prev)
    mix_softmax_kernel<<<(unsigned)nTok, 256, 0, stream>>>(
        cur, th_pre + i * TH_ * TH_, th_post + i * TH_ * TH_, at16);

    // o = attn @ v (v stored DK x S per (b,h) => transposed-B, contiguous)
    wmma_gemm_kernel<<<gemm_grid(TS, TDK, TB * TH_), 256, 0, stream>>>(
        at16, (long long)TS * TS, TS, v16, (long long)TDK * TS, TS, nullptr,
        nullptr, 0, 1.0f, nullptr, ob16, 0, TDK, TS, TDK, TS, 0, 2);

    // output projection (f32 out), then x = LN(x + o)
    wmma_gemm_kernel<<<gemm_grid((int)nTok, TD, 1), 256, 0, stream>>>(
        ob16, 0, TD, wo16 + iDD, 0, TD, bo + i * TD, nullptr, 0, 1.0f, op32,
        nullptr, 0, TD, (int)nTok, TD, TD, 0, 0);
    layernorm_kernel<<<(unsigned)nTok, 256, 0, stream>>>(
        op32, xr, ln1_g + i * TD, ln1_b + i * TD, x32, x16);
    xr = x32;

    // FFN: gelu(x @ W1 + b1) @ W2 + b2, then x = LN(x + ff)
    wmma_gemm_kernel<<<gemm_grid((int)nTok, TF, 1), 256, 0, stream>>>(
        x16, 0, TD, w1_16 + iDF, 0, TD, b1 + i * TF, nullptr, 0, 1.0f, nullptr,
        ff16, 0, TF, (int)nTok, TF, TD, 1, 0);
    wmma_gemm_kernel<<<gemm_grid((int)nTok, TD, 1), 256, 0, stream>>>(
        ff16, 0, TF, w2_16 + iDF, 0, TF, b2 + i * TD, nullptr, 0, 1.0f, ff32,
        nullptr, 0, TD, (int)nTok, TD, TF, 0, 0);
    float* xdst = (i == TL - 1) ? (float*)d_out : x32;
    layernorm_kernel<<<(unsigned)nTok, 256, 0, stream>>>(
        ff32, xr, ln2_g + i * TD, ln2_b + i * TD, xdst, x16);
    xr = xdst;
  }
}
